// MultiHeadSelfAttention_49426483642950
// MI455X (gfx1250) — matrix operand order repaired and checked by reading, not compiled, not run on hardware
//
#include <hip/hip_runtime.h>

#define B_  2
#define S_  2048
#define D_  1024
#define H_  16
#define HD_ 64

typedef __attribute__((ext_vector_type(16))) __bf16       bf16x16;
typedef __attribute__((ext_vector_type(8)))  float        f32x8;
typedef __attribute__((ext_vector_type(4)))  unsigned int u32x4;
typedef __attribute__((ext_vector_type(4)))  int          v4i;

typedef __attribute__((address_space(1))) v4i g_v4i;   // global int4
typedef __attribute__((address_space(3))) v4i l_v4i;   // LDS int4

#if defined(__gfx1250__) && __has_builtin(__builtin_amdgcn_global_load_async_to_lds_b128)
#define USE_ASYNC_COPY 1
#endif

union ABFrag {
    bf16x16 v;
    u32x4   q[2];
};

__device__ __forceinline__ unsigned short f2bf(float f) {
    unsigned int u = __float_as_uint(f);
    u += 0x7FFFu + ((u >> 16) & 1u);   // round-to-nearest-even
    return (unsigned short)(u >> 16);
}

__device__ __forceinline__ f32x8 wmma_bf16(bf16x16 a, bf16x16 b, f32x8 c) {
    return __builtin_amdgcn_wmma_f32_16x16x32_bf16(false, a, false, b, (short)0, c, false, false);
}

// copy 16 bytes (per lane) global -> LDS; async DMA path on CDNA5
__device__ __forceinline__ void async_copy16(void* lds, const void* glob) {
#ifdef USE_ASYNC_COPY
    __builtin_amdgcn_global_load_async_to_lds_b128(
        (g_v4i*)(unsigned long long)glob,
        (l_v4i*)(unsigned int)(unsigned long long)lds,
        0, 0);
#else
    *(u32x4*)lds = *(const u32x4*)glob;
#endif
}

__device__ __forceinline__ void async_wait_all() {
#ifdef USE_ASYNC_COPY
    asm volatile("s_wait_asynccnt 0x0" ::: "memory");
#endif
}

__device__ __forceinline__ float redmax16(float v) {
    v = fmaxf(v, __shfl_xor(v, 1));
    v = fmaxf(v, __shfl_xor(v, 2));
    v = fmaxf(v, __shfl_xor(v, 4));
    v = fmaxf(v, __shfl_xor(v, 8));
    return v;
}
__device__ __forceinline__ float redsum16(float v) {
    v += __shfl_xor(v, 1);
    v += __shfl_xor(v, 2);
    v += __shfl_xor(v, 4);
    v += __shfl_xor(v, 8);
    return v;
}

// ---------------------------------------------------------------- convert (plain)
__global__ __launch_bounds__(256) void mha_cvt_bf16(const float* __restrict__ src,
                                                    unsigned short* __restrict__ dst,
                                                    int n) {
    int i = blockIdx.x * blockDim.x + threadIdx.x;
    int stride = gridDim.x * blockDim.x;
    for (; i < n; i += stride) dst[i] = f2bf(src[i]);
}

// ---------------------------------------------------------------- convert + transpose
// src: rows x cols f32 row-major  ->  dst: cols x rows bf16 row-major
__global__ __launch_bounds__(256) void mha_cvt_bf16_t(const float* __restrict__ src,
                                                      unsigned short* __restrict__ dst,
                                                      int rows, int cols) {
    int n = rows * cols;
    int i = blockIdx.x * blockDim.x + threadIdx.x;
    int stride = gridDim.x * blockDim.x;
    for (; i < n; i += stride) {
        int r = i / cols;
        int c = i - r * cols;
        dst[(size_t)c * rows + r] = f2bf(src[i]);
    }
}

// ---------------------------------------------------------------- tile stagers
// A tile: 128 rows x 32 k, row stride D_ elements; dest [128][40]
__device__ __forceinline__ void gemm_issue(const unsigned short* __restrict__ A,
                                           const unsigned short* __restrict__ Bt,
                                           int mTile, int nTile, int k0,
                                           unsigned short (*Abuf)[40],
                                           unsigned short (*Bbuf)[40], int tid) {
#pragma unroll
    for (int it = 0; it < 2; ++it) {
        int cid = tid + it * 256;
        int row = cid >> 2;
        int off = (cid & 3) << 3;
        async_copy16(&Abuf[row][off], A + (size_t)(mTile + row) * D_ + k0 + off);
        async_copy16(&Bbuf[row][off], Bt + (size_t)(nTile + row) * D_ + k0 + off);
    }
}

// ---------------------------------------------------------------- QKV projection GEMM
// A: hBf (4096 x 1024); Bt: WqvkT (3072 x 1024), both bf16 row-major (Bt pre-transposed).
// N columns [0,1024)=Q, [1024,2048)=V, [2048,3072)=K (reference qvk order).
// Q,K out: [B][H][S][HD]; V out transposed: [B][H][HD][S].
__global__ __launch_bounds__(256) void mha_gemm_qkv(const unsigned short* __restrict__ A,
                                                    const unsigned short* __restrict__ Bt,
                                                    unsigned short* __restrict__ Qo,
                                                    unsigned short* __restrict__ Ko,
                                                    unsigned short* __restrict__ Vto) {
    __shared__ unsigned short Ash[2][128][40];
    __shared__ unsigned short Bsh[2][128][40];

    const int tid  = threadIdx.x;
    const int lane = tid & 31;
    const int wave = tid >> 5;
    const int ln   = lane & 15;
    const int half = lane >> 4;
    const int wm   = wave >> 2;   // 0..1
    const int wn   = wave & 3;    // 0..3
    const int mTile = blockIdx.x * 128;
    const int nTile = blockIdx.y * 128;

    f32x8 acc[4][2] = {};

    gemm_issue(A, Bt, mTile, nTile, 0, Ash[0], Bsh[0], tid);
    int bu = 0;
    for (int k0 = 0; k0 < D_; k0 += 32) {
        async_wait_all();
        __syncthreads();
        int nb = bu ^ 1;
        if (k0 + 32 < D_) gemm_issue(A, Bt, mTile, nTile, k0 + 32, Ash[nb], Bsh[nb], tid);

        ABFrag a[4], b[2];
#pragma unroll
        for (int i = 0; i < 4; ++i) {
            const unsigned short* p = &Ash[bu][wm * 64 + i * 16 + ln][8 * half];
            a[i].q[0] = *(const u32x4*)p;
            a[i].q[1] = *(const u32x4*)(p + 16);
        }
#pragma unroll
        for (int j = 0; j < 2; ++j) {
            const unsigned short* p = &Bsh[bu][wn * 32 + j * 16 + ln][8 * half];
            b[j].q[0] = *(const u32x4*)p;
            b[j].q[1] = *(const u32x4*)(p + 16);
        }
#pragma unroll
        for (int i = 0; i < 4; ++i)
#pragma unroll
            for (int j = 0; j < 2; ++j)
                acc[i][j] = wmma_bf16(a[i].v, b[j].v, acc[i][j]);
        bu = nb;
    }

#pragma unroll
    for (int i = 0; i < 4; ++i) {
#pragma unroll
        for (int j = 0; j < 2; ++j) {
            int gcol = nTile + wn * 32 + j * 16 + ln;  // 0..3071
            int blk  = gcol >> 10;                     // 0=q 1=v 2=k
            int d    = gcol & 1023;
            int head = d >> 6;
            int hd   = d & 63;
#pragma unroll
            for (int r = 0; r < 8; ++r) {
                int grow = mTile + wm * 64 + i * 16 + r + 8 * half;  // token
                int bb = grow >> 11;
                int ss = grow & (S_ - 1);
                unsigned short val = f2bf(acc[i][j][r]);
                if (blk == 0)
                    Qo[(((size_t)(bb * H_ + head) * S_ + ss) << 6) + hd] = val;
                else if (blk == 2)
                    Ko[(((size_t)(bb * H_ + head) * S_ + ss) << 6) + hd] = val;
                else
                    Vto[(((size_t)(bb * H_ + head) * HD_ + hd) << 11) + ss] = val;
            }
        }
    }
}

// ---------------------------------------------------------------- flash attention
// grid (qTile=16, head=16, batch=2); 8 waves, each owns 16 query rows.
// K: [B][H][S][HD], Vt: [B][H][HD][S] (pre-transposed) -> contiguous async tile copies.
__global__ __launch_bounds__(256) void mha_attn(const unsigned short* __restrict__ Q,
                                                const unsigned short* __restrict__ K,
                                                const unsigned short* __restrict__ Vt,
                                                unsigned short* __restrict__ WV) {
    __shared__ unsigned short Ksh[2][64][72];    // [key][hd]
    __shared__ unsigned short Vtsh[2][64][72];   // [hd][key]
    __shared__ unsigned short Psh[8][16][72];    // per-wave P scratch [m][k]

    const int tid  = threadIdx.x;
    const int lane = tid & 31;
    const int wave = tid >> 5;
    const int ln   = lane & 15;
    const int half = lane >> 4;

    const int bh        = blockIdx.z * H_ + blockIdx.y;
    const int qTileBase = blockIdx.x * 128;
    const unsigned short* Qh  = Q  + (size_t)bh * S_ * HD_;
    const unsigned short* Kh  = K  + (size_t)bh * S_ * HD_;
    const unsigned short* Vth = Vt + (size_t)bh * S_ * HD_;

    const int qRow = qTileBase + wave * 16;

    // Q fragments (16x64 = two 16x32 A-frags), loaded once
    ABFrag q0, q1;
    {
        const unsigned short* qp = Qh + (size_t)(qRow + ln) * HD_;
        q0.q[0] = *(const u32x4*)(qp + 8 * half);
        q0.q[1] = *(const u32x4*)(qp + 16 + 8 * half);
        q1.q[0] = *(const u32x4*)(qp + 32 + 8 * half);
        q1.q[1] = *(const u32x4*)(qp + 48 + 8 * half);
    }

    f32x8 o[4] = {};
    float mrow[8], lrow[8];
#pragma unroll
    for (int r = 0; r < 8; ++r) { mrow[r] = -3.0e38f; lrow[r] = 0.0f; }

    const int kEnd = qTileBase + 128;   // causal: no keys beyond query tile end

    // prologue: stage tile 0 into buffer 0
    for (int c = tid; c < 512; c += 256) {
        int row = c >> 3, off = (c & 7) << 3;
        async_copy16(&Ksh[0][row][off],  Kh  + (size_t)row * HD_ + off);
        async_copy16(&Vtsh[0][row][off], Vth + (size_t)row * S_ + off);
    }

    int bu = 0;
    for (int kt = 0; kt < kEnd; kt += 64) {
        async_wait_all();
        __syncthreads();
        int nb = bu ^ 1;
        if (kt + 64 < kEnd) {
            int knext = kt + 64;
            for (int c = tid; c < 512; c += 256) {
                int row = c >> 3, off = (c & 7) << 3;
                async_copy16(&Ksh[nb][row][off],  Kh  + (size_t)(knext + row) * HD_ + off);
                async_copy16(&Vtsh[nb][row][off], Vth + (size_t)row * S_ + knext + off);
            }
        }

        if (kt <= qRow + 15) {   // wave has at least one unmasked key here
            f32x8 s[4];
            float tmax[8];
#pragma unroll
            for (int r = 0; r < 8; ++r) tmax[r] = -3.0e38f;

            // scores: 16x64 = 4 sub-blocks of 16 keys, 2 WMMAs each
#pragma unroll
            for (int sub = 0; sub < 4; ++sub) {
                ABFrag kb0, kb1;
                const unsigned short* kp = &Ksh[bu][sub * 16 + ln][0];
                kb0.q[0] = *(const u32x4*)(kp + 8 * half);
                kb0.q[1] = *(const u32x4*)(kp + 16 + 8 * half);
                kb1.q[0] = *(const u32x4*)(kp + 32 + 8 * half);
                kb1.q[1] = *(const u32x4*)(kp + 48 + 8 * half);
                f32x8 sf = {};
                sf = wmma_bf16(q0.v, kb0.v, sf);
                sf = wmma_bf16(q1.v, kb1.v, sf);
                int key = kt + sub * 16 + ln;
#pragma unroll
                for (int r = 0; r < 8; ++r) {
                    int qi = qRow + r + 8 * half;
                    float val = sf[r] * 0.03125f;        // 1/sqrt(1024)
                    val = (key > qi) ? -3.0e38f : val;   // causal mask
                    sf[r] = val;
                    tmax[r] = fmaxf(tmax[r], val);
                }
                s[sub] = sf;
            }

            float alpha[8], tsum[8];
#pragma unroll
            for (int r = 0; r < 8; ++r) {
                tmax[r] = redmax16(tmax[r]);
                float mn = fmaxf(mrow[r], tmax[r]);
                alpha[r] = __expf(mrow[r] - mn);
                mrow[r]  = mn;
                tsum[r]  = 0.0f;
            }

            // P = exp(s - m) -> wave-private LDS in A-layout source form [m][k]
#pragma unroll
            for (int sub = 0; sub < 4; ++sub) {
#pragma unroll
                for (int r = 0; r < 8; ++r) {
                    float p = __expf(s[sub][r] - mrow[r]);
                    tsum[r] += p;
                    Psh[wave][r + 8 * half][sub * 16 + ln] = f2bf(p);
                }
            }
#pragma unroll
            for (int r = 0; r < 8; ++r) {
                tsum[r] = redsum16(tsum[r]);
                lrow[r] = lrow[r] * alpha[r] + tsum[r];
            }
#pragma unroll
            for (int j = 0; j < 4; ++j)
#pragma unroll
                for (int r = 0; r < 8; ++r) o[j][r] *= alpha[r];

            // P (A-frags) @ V (B-frags from transposed tile)
            ABFrag p0, p1;
            const unsigned short* pp = &Psh[wave][ln][0];
            p0.q[0] = *(const u32x4*)(pp + 8 * half);
            p0.q[1] = *(const u32x4*)(pp + 16 + 8 * half);
            p1.q[0] = *(const u32x4*)(pp + 32 + 8 * half);
            p1.q[1] = *(const u32x4*)(pp + 48 + 8 * half);
#pragma unroll
            for (int j = 0; j < 4; ++j) {
                ABFrag vb0, vb1;
                const unsigned short* vp = &Vtsh[bu][j * 16 + ln][0];
                vb0.q[0] = *(const u32x4*)(vp + 8 * half);
                vb0.q[1] = *(const u32x4*)(vp + 16 + 8 * half);
                vb1.q[0] = *(const u32x4*)(vp + 32 + 8 * half);
                vb1.q[1] = *(const u32x4*)(vp + 48 + 8 * half);
                o[j] = wmma_bf16(p0.v, vb0.v, o[j]);
                o[j] = wmma_bf16(p1.v, vb1.v, o[j]);
            }
        }
        bu = nb;
    }

    // normalize; write wv as [B][S][H*HD] bf16 (row-major input of out-proj)
#pragma unroll
    for (int j = 0; j < 4; ++j) {
        int col = blockIdx.y * HD_ + j * 16 + ln;
#pragma unroll
        for (int r = 0; r < 8; ++r) {
            int qi = qRow + r + 8 * half;
            float val = o[j][r] / lrow[r];
            WV[((size_t)(blockIdx.z * S_ + qi)) * D_ + col] = f2bf(val);
        }
    }
}

// ---------------------------------------------------------------- output projection GEMM
// A: wvBf (4096 x 1024); Bt: WoT (1024 x 1024, pre-transposed); out fp32 (4096 x 1024)
__global__ __launch_bounds__(256) void mha_gemm_out(const unsigned short* __restrict__ A,
                                                    const unsigned short* __restrict__ Bt,
                                                    float* __restrict__ out) {
    __shared__ unsigned short Ash[2][128][40];
    __shared__ unsigned short Bsh[2][128][40];

    const int tid  = threadIdx.x;
    const int lane = tid & 31;
    const int wave = tid >> 5;
    const int ln   = lane & 15;
    const int half = lane >> 4;
    const int wm   = wave >> 2;
    const int wn   = wave & 3;
    const int mTile = blockIdx.x * 128;
    const int nTile = blockIdx.y * 128;

    f32x8 acc[4][2] = {};

    gemm_issue(A, Bt, mTile, nTile, 0, Ash[0], Bsh[0], tid);
    int bu = 0;
    for (int k0 = 0; k0 < D_; k0 += 32) {
        async_wait_all();
        __syncthreads();
        int nb = bu ^ 1;
        if (k0 + 32 < D_) gemm_issue(A, Bt, mTile, nTile, k0 + 32, Ash[nb], Bsh[nb], tid);

        ABFrag a[4], b[2];
#pragma unroll
        for (int i = 0; i < 4; ++i) {
            const unsigned short* p = &Ash[bu][wm * 64 + i * 16 + ln][8 * half];
            a[i].q[0] = *(const u32x4*)p;
            a[i].q[1] = *(const u32x4*)(p + 16);
        }
#pragma unroll
        for (int j = 0; j < 2; ++j) {
            const unsigned short* p = &Bsh[bu][wn * 32 + j * 16 + ln][8 * half];
            b[j].q[0] = *(const u32x4*)p;
            b[j].q[1] = *(const u32x4*)(p + 16);
        }
#pragma unroll
        for (int i = 0; i < 4; ++i)
#pragma unroll
            for (int j = 0; j < 2; ++j)
                acc[i][j] = wmma_bf16(a[i].v, b[j].v, acc[i][j]);
        bu = nb;
    }

#pragma unroll
    for (int i = 0; i < 4; ++i) {
#pragma unroll
        for (int j = 0; j < 2; ++j) {
            int gcol = nTile + wn * 32 + j * 16 + ln;
#pragma unroll
            for (int r = 0; r < 8; ++r) {
                int grow = mTile + wm * 64 + i * 16 + r + 8 * half;
                out[(size_t)grow * D_ + gcol] = acc[i][j][r];
            }
        }
    }
}

// ---------------------------------------------------------------- launch
extern "C" void kernel_launch(void* const* d_in, const int* in_sizes, int n_in,
                              void* d_out, int out_size, void* d_ws, size_t ws_size,
                              hipStream_t stream) {
    (void)in_sizes; (void)n_in; (void)out_size;

    const float* h    = (const float*)d_in[0];
    const float* Wqvk = (const float*)d_in[1];
    const float* Wo   = (const float*)d_in[2];
    float* out = (float*)d_out;

    char* ws = (char*)d_ws;
    unsigned short* hBf    = (unsigned short*)(ws);              // 4096x1024
    unsigned short* wqvkT  = (unsigned short*)(ws + 8388608);    // 3072x1024 (transposed)
    unsigned short* woT    = (unsigned short*)(ws + 14680064);   // 1024x1024 (transposed)
    unsigned short* Qb     = (unsigned short*)(ws + 16777216);   // [B][H][S][HD]
    unsigned short* Kb     = (unsigned short*)(ws + 25165824);   // [B][H][S][HD]
    unsigned short* VbT    = (unsigned short*)(ws + 33554432);   // [B][H][HD][S]
    unsigned short* wvBf   = (unsigned short*)(ws + 41943040);   // 4096x1024
    if (ws_size < 50331648) return;

    mha_cvt_bf16<<<4096, 256, 0, stream>>>(h, hBf, B_ * S_ * D_);
    mha_cvt_bf16_t<<<4096, 256, 0, stream>>>(Wqvk, wqvkT, D_, 3 * D_);
    mha_cvt_bf16_t<<<2048, 256, 0, stream>>>(Wo, woT, D_, D_);

    mha_gemm_qkv<<<dim3(32, 24, 1), 256, 0, stream>>>(hBf, wqvkT, Qb, Kb, VbT);
    mha_attn<<<dim3(16, 16, 2), 256, 0, stream>>>(Qb, Kb, VbT, wvBf);
    mha_gemm_out<<<dim3(32, 8, 1), 256, 0, stream>>>(wvBf, woT, out);
}
